// NNUEModel_86053964743034
// MI455X (gfx1250) — compile-verified, matrix operand order checked by reading, not run on hardware
//
#include <hip/hip_runtime.h>

#define BATCH 65536
#define NEXP  8
#define FIN   640
#define FH    256
#define ROWS2 (2 * BATCH)          // 131072 routed rows (white + black)
#define TILE_M 64
#define MTILES (ROWS2 / TILE_M)    // 2048 (power of two)
#define KSTEPS (FIN / 32)          // 20
#define ABUF   2048                // elems per A staging buffer (4KB)

typedef __attribute__((ext_vector_type(16))) __bf16    v16bf;
typedef __attribute__((ext_vector_type(8)))  float     v8f;
typedef __attribute__((ext_vector_type(4)))  float     f32x4;   // native vectors:
typedef __attribute__((ext_vector_type(4)))  unsigned  u32x4;   // loadable from AS(1)

// global (address_space(1)) pointer helpers: force global_load_* instead of flat_*
#define GAS __attribute__((address_space(1)))
typedef const f32x4 GAS* gcf4;
typedef const u32x4 GAS* gcu4;
typedef u32x4       GAS* gu4;
template <typename T>
static __device__ __forceinline__ T as1(const void* p) {
  return (T)(unsigned long long)p;
}

// ---------------- workspace layout (bytes) ----------------
#define WS_COUNTS 0                              // 8 u32
#define WS_CURSOR 32                             // 8 u32
#define WS_OFFS   64                             // 8 u32
#define WS_PERM   128                            // ROWS2 * 4
#define WS_WB     (128 + ROWS2 * 4)              // E*H*IN bf16   (16B aligned)
#define WS_HCAT   (WS_WB + NEXP * FH * FIN * 2)  // B*512 bf16    (16B aligned)

// ---- fragment helpers ----
static __device__ __forceinline__ v16bf cvt_frag_f32(f32x4 x0, f32x4 x1,
                                                     f32x4 x2, f32x4 x3) {
  v16bf r;
  r[0]  = (__bf16)x0[0]; r[1]  = (__bf16)x0[1]; r[2]  = (__bf16)x0[2]; r[3]  = (__bf16)x0[3];
  r[4]  = (__bf16)x1[0]; r[5]  = (__bf16)x1[1]; r[6]  = (__bf16)x1[2]; r[7]  = (__bf16)x1[3];
  r[8]  = (__bf16)x2[0]; r[9]  = (__bf16)x2[1]; r[10] = (__bf16)x2[2]; r[11] = (__bf16)x2[3];
  r[12] = (__bf16)x3[0]; r[13] = (__bf16)x3[1]; r[14] = (__bf16)x3[2]; r[15] = (__bf16)x3[3];
  return r;
}
// f32 frag from global: elems p[0..7], p[16..23]  (q in f32x4 units)
static __device__ __forceinline__ v16bf load_frag_f32_g(gcf4 q) {
  return cvt_frag_f32(q[0], q[1], q[4], q[5]);
}
// bf16 frag from global: elems p[0..7], p[16..23]  (q in u32x4 units)
static __device__ __forceinline__ v16bf load_frag_bf16_g(gcu4 q) {
  union { u32x4 u[2]; v16bf v; } u;
  u.u[0] = q[0];
  u.u[1] = q[2];
  return u.v;
}
// bf16 frag from LDS: 16 contiguous elems (32B)
static __device__ __forceinline__ v16bf load_frag_bf16_lds(const __bf16* p) {
  union { u32x4 u[2]; v16bf v; } u;
  u.u[0] = *(const u32x4*)p;
  u.u[1] = *(const u32x4*)(p + 8);
  return u.v;
}
// convert 8 f32 -> 8 bf16 and store 16B to LDS
static __device__ __forceinline__ void store_cvt8(__bf16* d, f32x4 a, f32x4 b) {
  union { u32x4 q; __bf16 h[8]; } u;
  u.h[0] = (__bf16)a[0]; u.h[1] = (__bf16)a[1]; u.h[2] = (__bf16)a[2]; u.h[3] = (__bf16)a[3];
  u.h[4] = (__bf16)b[0]; u.h[5] = (__bf16)b[1]; u.h[6] = (__bf16)b[2]; u.h[7] = (__bf16)b[3];
  *(u32x4*)d = u.q;
}
static __device__ __forceinline__ v8f wmma_bf16(v16bf a, v16bf b, v8f c) {
  return __builtin_amdgcn_wmma_f32_16x16x32_bf16(false, a, false, b, (short)0, c, false, false);
}

// ---------------- routing kernels ----------------
__global__ void k_init(unsigned* ws_u32) {
  if (threadIdx.x < 16) ws_u32[threadIdx.x] = 0u;  // counts[8] + cursor[8]
}

__global__ void k_count(const int* __restrict__ wb, const int* __restrict__ bb,
                        unsigned* __restrict__ counts) {
  int r = blockIdx.x * 256 + threadIdx.x;          // r in [0, ROWS2)
  int bkt = ((r < BATCH) ? wb[r] : bb[r - BATCH]) & 7;
  atomicAdd(&counts[bkt], 1u);
}

__global__ void k_scan(const unsigned* __restrict__ counts, unsigned* __restrict__ cursor,
                       unsigned* __restrict__ offs) {
  if (threadIdx.x == 0 && blockIdx.x == 0) {
    unsigned s = 0;
    for (int e = 0; e < NEXP; ++e) { offs[e] = s; cursor[e] = s; s += counts[e]; }
  }
}

__global__ void k_scatter(const int* __restrict__ wb, const int* __restrict__ bb,
                          unsigned* __restrict__ cursor, int* __restrict__ perm) {
  int r = blockIdx.x * 256 + threadIdx.x;
  int bkt = ((r < BATCH) ? wb[r] : bb[r - BATCH]) & 7;
  unsigned p = atomicAdd(&cursor[bkt], 1u);
  perm[p] = r;
}

__global__ void k_cvtw(const float* __restrict__ Wft, __bf16* __restrict__ Wb) {
  int i = blockIdx.x * 256 + threadIdx.x;          // NEXP*FH*FIN threads
  Wb[i] = (__bf16)Wft[i];
}

// ---------------- expert feature-transform GEMM ----------------
// block: 256 threads (8 waves). Tile: 64 rows x 256 cols. K loop: 20 x K=32.
// A tile is gathered+converted cooperatively into LDS (double buffered, fragment
// order): producer is ds_store_b128, consumer is 2x ds_load_b128 per fragment,
// f32->bf16 conversion happens once per block instead of 8x.
__global__ __launch_bounds__(256) void k_ft(
    const float* __restrict__ wf, const float* __restrict__ bfeat,
    const int* __restrict__ perm, const unsigned* __restrict__ counts,
    const unsigned* __restrict__ offs, const __bf16* __restrict__ Wb,
    __bf16* __restrict__ hcat) {
  const int e = blockIdx.x >> 11;               // MTILES == 2048
  const int t = blockIdx.x & (MTILES - 1);
  const unsigned cnt = counts[e];
  if ((unsigned)(t * TILE_M) >= cnt) return;
  const unsigned off = offs[e];

  const int tid  = threadIdx.x;
  const int wave = tid >> 5;
  const int lane = tid & 31;
  const int lrow = lane & 15;
  const int hi   = lane >> 4;                   // 0: K{0-7,16-23}; 1: K{8-15,24-31}

  // A staging: [buf][mt(4)][lane(32)][elem(16)] bf16 = 4KB per buffer
  __shared__ __align__(16) __bf16 ldsA[2 * ABUF];

  // ---- producer setup: thread = (row 0..63, k-quarter 0..3) ----
  const int prow = tid >> 2;
  const int g    = tid & 3;                     // which 8-float group of the 32-K step
  unsigned pli = (unsigned)(t * TILE_M + prow);
  unsigned plc = pli < cnt ? pli : cnt - 1u;    // clamp (stores masked later)
  const int pr = perm[off + plc];
  const float* fbase = (pr < BATCH) ? (wf + (size_t)pr * FIN)
                                    : (bfeat + (size_t)(pr - BATCH) * FIN);
  gcf4 pA = as1<gcf4>(fbase + g * 8);           // indexed in f32x4; +8 per K step
  // LDS destination (fragment order); ping-pong selected by INTEGER offset so the
  // pointer stays in addrspace(3) -> ds_store_b128 (not flat_store).
  // g=0 -> lane lrow elems 0-7 (K0-7);  g=2 -> lane lrow elems 8-15 (K16-23)
  // g=1 -> lane lrow+16 elems 0-7;      g=3 -> lane lrow+16 elems 8-15
  const int mt_p  = prow >> 4, lrow_p = prow & 15;
  const int dlane = lrow_p + (g & 1) * 16;
  const int ebase = (g >> 1) * 8;
  __bf16* pdst = &ldsA[mt_p * 512 + dlane * 16 + ebase];

  // ---- B (weights, bf16) per-wave pointers ----
  const __bf16* wbase = Wb + (size_t)e * FH * FIN;
  const int nb = wave * 32;
  gcu4 pB[2];
#pragma unroll
  for (int nt = 0; nt < 2; ++nt)
    pB[nt] = as1<gcu4>(wbase + (size_t)(nb + nt * 16 + lrow) * FIN + hi * 8);

  v8f acc[4][2] = {};

  // prologue: fill buffer 0 with K step 0
  {
    f32x4 f0 = pA[0], f1 = pA[1];
    store_cvt8(pdst, f0, f1);
  }
  __syncthreads();

  for (int ks = 0; ks < KSTEPS; ++ks) {
    const int cur = ks & 1;
    const bool more = (ks + 1 < KSTEPS);
    // issue next K-step's global A loads early (overlap with WMMAs)
    f32x4 n0, n1;
    if (more) { n0 = pA[(ks + 1) * 8]; n1 = pA[(ks + 1) * 8 + 1]; }
    // B fragments for this step
    v16bf bfr[2];
#pragma unroll
    for (int nt = 0; nt < 2; ++nt) bfr[nt] = load_frag_bf16_g(pB[nt] + ks * 4);
    // A fragments from LDS (integer buffer offset keeps DS addressing)
    const __bf16* asrc = &ldsA[cur * ABUF + lane * 16];
    v16bf afr[4];
#pragma unroll
    for (int mt = 0; mt < 4; ++mt) afr[mt] = load_frag_bf16_lds(asrc + mt * 512);
#pragma unroll
    for (int mt = 0; mt < 4; ++mt)
#pragma unroll
      for (int nt = 0; nt < 2; ++nt) acc[mt][nt] = wmma_bf16(afr[mt], bfr[nt], acc[mt][nt]);
    if (more) store_cvt8(pdst + (cur ^ 1) * ABUF, n0, n1);
    __syncthreads();
  }

  // relu -> LDS staging (bf16, row-major 64x256)
  __shared__ __align__(16) __bf16 ldsO[TILE_M * FH];
#pragma unroll
  for (int mt = 0; mt < 4; ++mt)
#pragma unroll
    for (int nt = 0; nt < 2; ++nt)
#pragma unroll
      for (int v = 0; v < 8; ++v) {
        float x = fmaxf(acc[mt][nt][v], 0.0f);
        int m = mt * 16 + v + hi * 8;
        int n = nb + nt * 16 + lrow;
        ldsO[m * FH + n] = (__bf16)x;
      }
  __syncthreads();

  // cooperative vector scatter: 4 threads per row, 64 cols (128B) each
  const int row = tid >> 2, seg = tid & 3;
  unsigned li = (unsigned)(t * TILE_M + row);
  if (li < cnt) {
    int r = perm[off + li];
    int b = (r < BATCH) ? r : r - BATCH;
    int colbase = (r < BATCH) ? 0 : FH;
    gu4 d4 = as1<gu4>(hcat + (size_t)b * (2 * FH) + colbase + seg * 64);
    const u32x4* s4 = (const u32x4*)(ldsO + row * FH + seg * 64);
#pragma unroll
    for (int i = 0; i < 8; ++i) d4[i] = s4[i];
  }
}

// ---------------- output MLP ----------------
// 128 threads (4 waves); each wave handles 16 samples.
__global__ __launch_bounds__(128) void k_mlp(
    const __bf16* __restrict__ hcat,
    const float* __restrict__ W1, const float* __restrict__ b1,
    const float* __restrict__ W2, const float* __restrict__ b2,
    const float* __restrict__ W3, const float* __restrict__ b3,
    const float* __restrict__ stm, float* __restrict__ out) {
  const int tid = threadIdx.x, wave = tid >> 5, lane = tid & 31;
  const int lrow = lane & 15, hi = lane >> 4;
  const int s0 = (blockIdx.x * 4 + wave) * 16;

  __shared__ __align__(16) __bf16 l1s[4][16 * 32];
  __shared__ float                l2s[4][16 * 32];

  // layer 1: [16x512] x [512x32]
  const __bf16* aptr = hcat + (size_t)(s0 + lrow) * 512 + hi * 8;
  v8f acc1[2] = {};
  for (int k = 0; k < 512; k += 32) {
    v16bf afr = load_frag_bf16_g(as1<gcu4>(aptr + k));
#pragma unroll
    for (int nt = 0; nt < 2; ++nt) {
      const float* wp = W1 + (size_t)(nt * 16 + lrow) * 512 + k + hi * 8;
      acc1[nt] = wmma_bf16(afr, load_frag_f32_g(as1<gcf4>(wp)), acc1[nt]);
    }
  }
#pragma unroll
  for (int nt = 0; nt < 2; ++nt) {
    float bias = b1[nt * 16 + lrow];
#pragma unroll
    for (int v = 0; v < 8; ++v) {
      float x = fmaxf(acc1[nt][v] + bias, 0.0f);
      l1s[wave][(v + hi * 8) * 32 + nt * 16 + lrow] = (__bf16)x;
    }
  }
  __syncthreads();

  // layer 2: [16x32] x [32x32] — exactly one K=32 WMMA step
  v16bf a2;
  {
    union { u32x4 u[2]; v16bf v; } u;
    const __bf16* p = &l1s[wave][lrow * 32 + hi * 8];
    u.u[0] = *(const u32x4*)p;          // elems {K0-7 | K8-15}
    u.u[1] = *(const u32x4*)(p + 16);   // elems {K16-23 | K24-31}
    a2 = u.v;
  }
  v8f acc2[2] = {};
#pragma unroll
  for (int nt = 0; nt < 2; ++nt) {
    const float* wp = W2 + (size_t)(nt * 16 + lrow) * 32 + hi * 8;
    acc2[nt] = wmma_bf16(a2, load_frag_f32_g(as1<gcf4>(wp)), acc2[nt]);
  }
#pragma unroll
  for (int nt = 0; nt < 2; ++nt) {
    float bias = b2[nt * 16 + lrow];
#pragma unroll
    for (int v = 0; v < 8; ++v)
      l2s[wave][(v + hi * 8) * 32 + nt * 16 + lrow] = fmaxf(acc2[nt][v] + bias, 0.0f);
  }
  __syncthreads();

  // layer 3: [16x32] x [32x1] + stm scale (VALU, lanes 0..15)
  if (lane < 16) {
    int s = s0 + lane;
    float a3 = b3[0];
#pragma unroll
    for (int j = 0; j < 32; ++j) a3 += l2s[wave][lane * 32 + j] * W3[j];
    out[s] = a3 * stm[s];
  }
}

// ---------------- host launcher ----------------
extern "C" void kernel_launch(void* const* d_in, const int* in_sizes, int n_in,
                              void* d_out, int out_size, void* d_ws, size_t ws_size,
                              hipStream_t stream) {
  (void)in_sizes; (void)n_in; (void)out_size; (void)ws_size;
  const float* wf   = (const float*)d_in[0];
  const float* bft  = (const float*)d_in[1];
  const int*   wbk  = (const int*)d_in[2];
  const int*   bbk  = (const int*)d_in[3];
  const float* stm  = (const float*)d_in[4];
  const float* Wft  = (const float*)d_in[5];
  const float* W1   = (const float*)d_in[6];
  const float* b1   = (const float*)d_in[7];
  const float* W2   = (const float*)d_in[8];
  const float* b2   = (const float*)d_in[9];
  const float* W3   = (const float*)d_in[10];
  const float* b3   = (const float*)d_in[11];
  float* out = (float*)d_out;

  char* ws = (char*)d_ws;
  unsigned* counts = (unsigned*)(ws + WS_COUNTS);
  unsigned* cursor = (unsigned*)(ws + WS_CURSOR);
  unsigned* offs   = (unsigned*)(ws + WS_OFFS);
  int*      perm   = (int*)(ws + WS_PERM);
  __bf16*   Wb     = (__bf16*)(ws + WS_WB);
  __bf16*   hcat   = (__bf16*)(ws + WS_HCAT);

  k_init<<<1, 32, 0, stream>>>((unsigned*)ws);
  k_count<<<ROWS2 / 256, 256, 0, stream>>>(wbk, bbk, counts);
  k_scan<<<1, 32, 0, stream>>>(counts, cursor, offs);
  k_scatter<<<ROWS2 / 256, 256, 0, stream>>>(wbk, bbk, cursor, perm);
  k_cvtw<<<(NEXP * FH * FIN) / 256, 256, 0, stream>>>(Wft, Wb);
  k_ft<<<NEXP * MTILES, 256, 0, stream>>>(wf, bft, perm, counts, offs, Wb, hcat);
  k_mlp<<<BATCH / 64, 128, 0, stream>>>(hcat, W1, b1, W2, b2, W3, b3, stm, out);
}
